// ModAFNO_49435073577170
// MI455X (gfx1250) — compile-verified, matrix operand order checked by reading, not run on hardware
//
#include <hip/hip_runtime.h>
#include <hip/hip_bf16.h>

// ---------------------------------------------------------------------------
// ModAFNO forward for MI455X (gfx1250).  Everything heavy is mapped onto
// v_wmma_f32_16x16x32_f16: patch-embed (im2col GEMM), rfft2/irfft2 as DFT
// matmuls, block-diagonal complex spectral MLP as batched 32x32 GEMMs,
// token MLP and head.  Elementwise kernels handle layernorm / modulation /
// softshrink / residuals.
// ---------------------------------------------------------------------------

typedef __attribute__((ext_vector_type(16))) _Float16 v16h;
typedef __attribute__((ext_vector_type(8)))  float    v8f;
// 4-byte-aligned float4 (some A matrices have lda % 4 != 0, e.g. 181)
typedef float v4fu __attribute__((ext_vector_type(4), aligned(4)));

#define H_IMG 360
#define W_IMG 720
#define HP 180
#define WP 360
#define NTOK (HP * WP)          // 64800
#define C_IN 155
#define C_OUT 73
#define EMBED 256
#define KPATCH (C_IN * 4)       // 620
#define LATENT 512
#define WF 181                  // rfft width bins
#define SPEC_P (HP * WF)        // 32580 spectral positions
#define SPEC_ELEMS ((size_t)SPEC_P * EMBED)
#define LAMBDA_SS 0.01f
#define LDS_S 72                // LDS row stride in halfs (64-wide K tile + pad)

__device__ __forceinline__ float geluf(float x) {
    float x3 = x * x * x;
    return 0.5f * x * (1.0f + tanhf(0.7978845608028654f * (x + 0.044715f * x3)));
}

// ---------------------------------------------------------------------------
// Generic batched GEMM:  C[z] (MxN f32) = alpha * A[z](MxK) * B[z](KxN)
//                                        + bias[n] + beta * C[z]  (opt GELU)
// A/B are f32 in global, converted to f16 while staging into LDS.
// Tile 64x64, K-tile 64 (8 WMMAs per stage), double-buffered LDS so the
// staging loads for tile t+1 overlap the WMMA burst of tile t (one barrier
// per K iteration).  Interior tiles use unrolled b128 loads; edge tiles use
// branch-free clamped-address loads + zero select.
// ---------------------------------------------------------------------------
__launch_bounds__(128)
__global__ void gemm_wmma_kernel(const float* __restrict__ A,
                                 const float* __restrict__ B,
                                 float* __restrict__ C,
                                 const float* __restrict__ bias,
                                 int M, int N, int K,
                                 int lda, int ldb, int ldc,
                                 long long sA, long long sB, long long sC,
                                 float alpha, float beta, int act)
{
    __shared__ _Float16 ldsA[2][64 * LDS_S];   // [m][k]
    __shared__ _Float16 ldsB[2][64 * LDS_S];   // [n][k] (transposed)

    const int tid   = threadIdx.x;
    const int lane  = tid & 31;
    const int wave  = tid >> 5;
    const int hi    = lane >> 4;       // half-wave select
    const int l     = lane & 15;
    const int mBase = blockIdx.y * 64;
    const int nBase = blockIdx.x * 64;
    const long long z = blockIdx.z;

    A += z * sA;
    B += z * sB;
    C += z * sC;

    const bool mFull = (mBase + 64 <= M);
    const bool nFull = (nBase + 64 <= N);

    v8f acc[4] = {};

    // ---- stage one 64x64 K-tile of A and B into LDS buffers ----
    auto stage = [&](int kt, _Float16* dA, _Float16* dB) {
        const bool kFull = (kt + 64 <= K);
        if (mFull && kFull) {
#pragma unroll
            for (int t = 0; t < 8; ++t) {
                int idx4 = tid + t * 128;          // 1024 float4s
                int i = idx4 >> 4;                 // 16 float4 per row
                int j = (idx4 & 15) * 4;
                v4fu v = *(const v4fu*)&A[(long long)(mBase + i) * lda + kt + j];
                _Float16* d = &dA[i * LDS_S + j];
                d[0] = (_Float16)v.x; d[1] = (_Float16)v.y;
                d[2] = (_Float16)v.z; d[3] = (_Float16)v.w;
            }
        } else {
            // branch-free: clamp address into range, select 0 for OOB
#pragma unroll 4
            for (int t = 0; t < 32; ++t) {
                int idx = tid + t * 128;
                int i = idx >> 6, j = idx & 63;
                int gm = mBase + i, gk = kt + j;
                int cm = gm < M ? gm : M - 1;
                int ck = gk < K ? gk : K - 1;
                float v = A[(long long)cm * lda + ck];
                v = (gm < M && gk < K) ? v : 0.0f;
                dA[i * LDS_S + j] = (_Float16)v;
            }
        }
        if (nFull && kFull) {
#pragma unroll
            for (int t = 0; t < 8; ++t) {
                int idx4 = tid + t * 128;          // 1024 float4s
                int k = idx4 >> 4;                 // 16 float4 per row
                int n = (idx4 & 15) * 4;
                v4fu v = *(const v4fu*)&B[(long long)(kt + k) * ldb + nBase + n];
                dB[(n + 0) * LDS_S + k] = (_Float16)v.x;
                dB[(n + 1) * LDS_S + k] = (_Float16)v.y;
                dB[(n + 2) * LDS_S + k] = (_Float16)v.z;
                dB[(n + 3) * LDS_S + k] = (_Float16)v.w;
            }
        } else {
#pragma unroll 4
            for (int t = 0; t < 32; ++t) {
                int idx = tid + t * 128;
                int k = idx >> 6, n = idx & 63;
                int gk = kt + k, gn = nBase + n;
                int cgk = gk < K ? gk : K - 1;
                int cgn = gn < N ? gn : N - 1;
                float v = B[(long long)cgk * ldb + cgn];
                v = (gk < K && gn < N) ? v : 0.0f;
                dB[n * LDS_S + k] = (_Float16)v;
            }
        }
    };

    // ---- 8 WMMAs from one staged K-tile ----
    auto compute = [&](int kt, const _Float16* pA, const _Float16* pB) {
#pragma unroll
        for (int ko = 0; ko < 2; ++ko) {
            if (kt + ko * 32 < K) {   // skip empty second half for K<=32 tiles
                // A fragment: 16x32, lanes 0-15 K[0..7,16..23], 16-31 K[8..15,24..31]
                v16h af;
                const _Float16* pa = &pA[(wave * 16 + l) * LDS_S + ko * 32];
#pragma unroll
                for (int p = 0; p < 8; ++p) {
                    int k = ((p & 3) << 1) + ((p >> 2) << 4) + (hi << 3);
                    af[2 * p]     = pa[k];
                    af[2 * p + 1] = pa[k + 1];
                }
                // preload all four B fragments, then issue WMMAs back-to-back
                v16h bf[4];
#pragma unroll
                for (int nb = 0; nb < 4; ++nb) {
                    const _Float16* pb = &pB[(nb * 16 + l) * LDS_S + ko * 32];
#pragma unroll
                    for (int p = 0; p < 8; ++p) {
                        int k = ((p & 3) << 1) + ((p >> 2) << 4) + (hi << 3);
                        bf[nb][2 * p]     = pb[k];
                        bf[nb][2 * p + 1] = pb[k + 1];
                    }
                }
#pragma unroll
                for (int nb = 0; nb < 4; ++nb) {
                    acc[nb] = __builtin_amdgcn_wmma_f32_16x16x32_f16(
                        false, af, false, bf[nb], (short)0, acc[nb], false, false);
                }
            }
        }
    };

    const int nk = (K + 63) >> 6;
    stage(0, ldsA[0], ldsB[0]);
    __syncthreads();
    for (int t = 0; t < nk; ++t) {
        const int kt  = t << 6;
        const int cur = t & 1;
        if (t + 1 < nk) {
            stage(kt + 64, ldsA[cur ^ 1], ldsB[cur ^ 1]);
            if (t + 2 < nk) {   // prefetch tile t+2 (global_prefetch_b8)
                int pm = mBase + (tid & 63);
                if (pm < M) __builtin_prefetch(&A[(long long)pm * lda + kt + 128], 0, 1);
                int pk = kt + 128 + (tid & 63);
                if (pk < K) __builtin_prefetch(&B[(long long)pk * ldb + nBase], 0, 1);
            }
        }
        compute(kt, ldsA[cur], ldsB[cur]);
        __syncthreads();
    }

    // ---- epilogue: C layout VGPR r -> M = r + hi*8, N = lane&15 ----
    if (mFull && nFull) {
#pragma unroll
        for (int nb = 0; nb < 4; ++nb) {
            int n = nBase + nb * 16 + l;
            float bv = bias ? bias[n] : 0.0f;
#pragma unroll
            for (int r = 0; r < 8; ++r) {
                int m = mBase + wave * 16 + r + hi * 8;
                float v = alpha * acc[nb][r] + bv;
                if (beta != 0.0f) v += beta * C[(long long)m * ldc + n];
                if (act) v = geluf(v);
                C[(long long)m * ldc + n] = v;
            }
        }
    } else {
#pragma unroll
        for (int nb = 0; nb < 4; ++nb) {
            int n = nBase + nb * 16 + l;
            if (n >= N) continue;
            float bv = bias ? bias[n] : 0.0f;
#pragma unroll
            for (int r = 0; r < 8; ++r) {
                int m = mBase + wave * 16 + r + hi * 8;
                if (m >= M) continue;
                float v = alpha * acc[nb][r] + bv;
                if (beta != 0.0f) v += beta * C[(long long)m * ldc + n];
                if (act) v = geluf(v);
                C[(long long)m * ldc + n] = v;
            }
        }
    }
}

// ---------------------------------------------------------------------------
// Elementwise / small kernels
// ---------------------------------------------------------------------------

// DFT twiddles, forward along W (rfft, 181 x 360), ortho scaled
__global__ void gen_wfwd_kernel(float* tc, float* ts) {
    int idx = blockIdx.x * blockDim.x + threadIdx.x;
    if (idx >= WF * WP) return;
    int wf = idx / WP, w = idx % WP;
    float th = 6.283185307179586f * (float)w * (float)wf / (float)WP;
    float s = rsqrtf((float)WP);
    tc[idx] = cosf(th) * s;
    ts[idx] = -sinf(th) * s;
}

// DFT twiddles along H (180x180), sign=-1 fwd, +1 inverse, ortho scaled
__global__ void gen_dfth_kernel(float* tc, float* ts, float sign) {
    int idx = blockIdx.x * blockDim.x + threadIdx.x;
    if (idx >= HP * HP) return;
    int a = idx / HP, b = idx % HP;
    float th = 6.283185307179586f * (float)a * (float)b / (float)HP;
    float s = rsqrtf((float)HP);
    tc[idx] = cosf(th) * s;
    ts[idx] = sign * sinf(th) * s;
}

// inverse real DFT along W (360 x 181) with hermitian doubling
__global__ void gen_winv_kernel(float* ur, float* ui) {
    int idx = blockIdx.x * blockDim.x + threadIdx.x;
    if (idx >= WP * WF) return;
    int w = idx / WF, wf = idx % WF;
    float alpha = (wf == 0 || wf == WP / 2) ? 1.0f : 2.0f;
    float th = 6.283185307179586f * (float)w * (float)wf / (float)WP;
    float s = rsqrtf((float)WP);
    ur[idx] = alpha * cosf(th) * s;
    ui[idx] = -alpha * sinf(th) * s;
}

// sinusoidal time embedding: [sin(mod*2pi*k), cos(mod*2pi*k)], k=1..32
__global__ void mod_sincos_kernel(const float* mod, float* emb) {
    int i = threadIdx.x;
    if (i < 32) {
        float a = mod[0] * 6.283185307179586f * (float)(i + 1);
        emb[i] = sinf(a);
        emb[i + 32] = cosf(a);
    }
}

// small dense matvec: out[n] = act(sum_k in[k]*W[k*N+n] + b[n])
__global__ void matvec_kernel(const float* __restrict__ W, const float* __restrict__ b,
                              const float* __restrict__ in, float* __restrict__ out,
                              int K, int N, int act) {
    int n = blockIdx.x * blockDim.x + threadIdx.x;
    if (n >= N) return;
    float s = b ? b[n] : 0.0f;
    for (int k = 0; k < K; ++k) s += in[k] * W[k * N + n];
    out[n] = act ? geluf(s) : s;
}

// im2col for 2x2 stride-2 patches of NCHW input
__global__ void im2col_kernel(const float* __restrict__ x, float* __restrict__ A) {
    long long idx = (long long)blockIdx.x * blockDim.x + threadIdx.x;
    if (idx >= (long long)NTOK * KPATCH) return;
    int n = (int)(idx / KPATCH), kk = (int)(idx % KPATCH);
    int c = kk >> 2, p = kk & 3, p1 = p >> 1, p2 = p & 1;
    int hp = n / WP, wp = n % WP;
    A[idx] = x[((long long)c * H_IMG + (hp * 2 + p1)) * W_IMG + (wp * 2 + p2)];
}

// transpose patch weight (256,155,2,2) -> (620,256)
__global__ void transpose_w_kernel(const float* __restrict__ w, float* __restrict__ wt) {
    int idx = blockIdx.x * blockDim.x + threadIdx.x;
    if (idx >= KPATCH * EMBED) return;
    int kk = idx / EMBED, e = idx % EMBED;
    wt[idx] = w[e * KPATCH + kk];
}

// xt += pos + modvec (broadcast over tokens)
__global__ void add_pos_mod_kernel(float* __restrict__ xt, const float* __restrict__ pos,
                                   const float* __restrict__ mv) {
    long long idx = (long long)blockIdx.x * blockDim.x + threadIdx.x;
    if (idx >= (long long)NTOK * EMBED) return;
    xt[idx] += pos[idx] + mv[idx & (EMBED - 1)];
}

// layernorm over 256 channels, one wave32 per row; optional mss scale/shift
__global__ void layernorm_kernel(const float* __restrict__ x, const float* __restrict__ g,
                                 const float* __restrict__ b, const float* __restrict__ scsh,
                                 float* __restrict__ out, int rows) {
    int row = blockIdx.x * 8 + (threadIdx.x >> 5);
    int lane = threadIdx.x & 31;
    if (row >= rows) return;
    const float* xr = x + (long long)row * EMBED;
    float vals[8];
    float s = 0.0f;
#pragma unroll
    for (int i = 0; i < 8; ++i) { vals[i] = xr[lane + i * 32]; s += vals[i]; }
#pragma unroll
    for (int off = 16; off; off >>= 1) s += __shfl_xor(s, off, 32);
    float mean = s * (1.0f / EMBED);
    float v = 0.0f;
#pragma unroll
    for (int i = 0; i < 8; ++i) { float d = vals[i] - mean; v += d * d; }
#pragma unroll
    for (int off = 16; off; off >>= 1) v += __shfl_xor(v, off, 32);
    float rs = rsqrtf(v * (1.0f / EMBED) + 1e-6f);
#pragma unroll
    for (int i = 0; i < 8; ++i) {
        int c = lane + i * 32;
        float o = (vals[i] - mean) * rs * g[c] + b[c];
        if (scsh) o = o * (scsh[c] + 1.0f) + scsh[EMBED + c];
        out[(long long)row * EMBED + c] = o;
    }
}

// spectral layer1 epilogue: relu(x + b1) then complex modulation scale/shift
__global__ void spectral_mod_kernel(float* __restrict__ o1r, float* __restrict__ o1i,
                                    const float* __restrict__ b1, const float* __restrict__ fss) {
    long long idx = (long long)blockIdx.x * blockDim.x + threadIdx.x;
    if (idx >= (long long)SPEC_ELEMS) return;
    int c = (int)(idx & (EMBED - 1));
    float r = fmaxf(o1r[idx] + b1[c], 0.0f);
    float im = fmaxf(o1i[idx] + b1[EMBED + c], 0.0f);
    float sre = fss[c] + 1.0f, sim = fss[EMBED + c] + 1.0f;
    float shr = fss[2 * EMBED + c], shi = fss[3 * EMBED + c];
    o1r[idx] = sre * r - sim * im + shr;
    o1i[idx] = sim * r + sre * im + shi;
}

// spectral layer2 epilogue: softshrink(x + b2)
__global__ void softshrink_kernel(float* __restrict__ o2r, float* __restrict__ o2i,
                                  const float* __restrict__ b2) {
    long long idx = (long long)blockIdx.x * blockDim.x + threadIdx.x;
    if (idx >= (long long)SPEC_ELEMS) return;
    int c = (int)(idx & (EMBED - 1));
    float r = o2r[idx] + b2[c];
    float im = o2i[idx] + b2[EMBED + c];
    float sr = copysignf(fmaxf(fabsf(r) - LAMBDA_SS, 0.0f), r);
    float si = copysignf(fmaxf(fabsf(im) - LAMBDA_SS, 0.0f), im);
    o2r[idx] = sr;
    o2i[idx] = si;
}

// xt += a + b   (AFNO residual: irfft_out + filter input + outer residual)
__global__ void add3_kernel(float* __restrict__ xt, const float* __restrict__ a,
                            const float* __restrict__ b) {
    long long idx = (long long)blockIdx.x * blockDim.x + threadIdx.x;
    if (idx >= (long long)NTOK * EMBED) return;
    xt[idx] += a[idx] + b[idx];
}

// unpatchify head output (Hp,Wp,2,2,73) -> (73,360,720)
__global__ void unpatch_kernel(const float* __restrict__ y, float* __restrict__ out) {
    long long idx = (long long)blockIdx.x * blockDim.x + threadIdx.x;
    if (idx >= (long long)C_OUT * H_IMG * W_IMG) return;
    int co = (int)(idx / (H_IMG * W_IMG));
    int rem = (int)(idx % (H_IMG * W_IMG));
    int h = rem / W_IMG, w = rem % W_IMG;
    int hp = h >> 1, p1 = h & 1, wp = w >> 1, p2 = w & 1;
    out[idx] = y[((long long)(hp * WP + wp)) * (4 * C_OUT) + (p1 * 2 + p2) * C_OUT + co];
}

// ---------------------------------------------------------------------------
// Host orchestration
// ---------------------------------------------------------------------------
extern "C" void kernel_launch(void* const* d_in, const int* in_sizes, int n_in,
                              void* d_out, int out_size, void* d_ws, size_t ws_size,
                              hipStream_t stream)
{
    (void)in_sizes; (void)n_in; (void)out_size;

    const float* x         = (const float*)d_in[0];
    const float* mod       = (const float*)d_in[1];
    const float* patch_w   = (const float*)d_in[2];
    const float* patch_b   = (const float*)d_in[3];
    const float* pos       = (const float*)d_in[4];
    const float* emb_w     = (const float*)d_in[5];
    const float* emb_b     = (const float*)d_in[6];
    const float* modproj_w = (const float*)d_in[7];
    const float* modproj_b = (const float*)d_in[8];

    struct Blk {
        const float *n1g, *n1b, *w1, *b1, *w2, *b2;
        const float *fw1, *fb1, *fw2, *fb2;
        const float *n2g, *n2b, *mw1, *mb1, *mw2, *mb2;
        const float *sw1, *sb1, *sw2, *sb2;
    };
    Blk blk[2];
    int bi = 9;
    for (int b = 0; b < 2; ++b) {
        blk[b].n1g = (const float*)d_in[bi + 0];  blk[b].n1b = (const float*)d_in[bi + 1];
        blk[b].w1  = (const float*)d_in[bi + 2];  blk[b].b1  = (const float*)d_in[bi + 3];
        blk[b].w2  = (const float*)d_in[bi + 4];  blk[b].b2  = (const float*)d_in[bi + 5];
        blk[b].fw1 = (const float*)d_in[bi + 6];  blk[b].fb1 = (const float*)d_in[bi + 7];
        blk[b].fw2 = (const float*)d_in[bi + 8];  blk[b].fb2 = (const float*)d_in[bi + 9];
        blk[b].n2g = (const float*)d_in[bi + 10]; blk[b].n2b = (const float*)d_in[bi + 11];
        blk[b].mw1 = (const float*)d_in[bi + 12]; blk[b].mb1 = (const float*)d_in[bi + 13];
        blk[b].mw2 = (const float*)d_in[bi + 14]; blk[b].mb2 = (const float*)d_in[bi + 15];
        blk[b].sw1 = (const float*)d_in[bi + 16]; blk[b].sb1 = (const float*)d_in[bi + 17];
        blk[b].sw2 = (const float*)d_in[bi + 18]; blk[b].sb2 = (const float*)d_in[bi + 19];
        bi += 20;
    }
    const float* head_w = (const float*)d_in[bi];

    // ---- workspace arena (floats) ----
    float* W = (float*)d_ws;
    size_t off = 0;
    auto alloc = [&](size_t n) { float* p = W + off; off += n; return p; };

    float* R0   = alloc((size_t)NTOK * KPATCH);   // im2col, then spectral ping-pong
    float* BW   = alloc((size_t)KPATCH * EMBED);
    float* XT   = alloc((size_t)NTOK * EMBED);
    float* LN   = alloc((size_t)NTOK * EMBED);
    float* HID  = alloc((size_t)NTOK * LATENT);   // mlp hidden / irfft temp
    float* HEAD = alloc((size_t)NTOK * 4 * C_OUT);
    float* TWc  = alloc((size_t)WF * WP);
    float* TWs  = alloc((size_t)WF * WP);
    float* Fc   = alloc((size_t)HP * HP);
    float* Fs   = alloc((size_t)HP * HP);
    float* Gc   = alloc((size_t)HP * HP);
    float* Gs   = alloc((size_t)HP * HP);
    float* Ur   = alloc((size_t)WP * WF);
    float* Ui   = alloc((size_t)WP * WF);
    float* EMB  = alloc(64);
    float* ME   = alloc(64);
    float* MV   = alloc(EMBED);
    float* FH   = alloc(1024);
    float* FO0  = alloc(1024);
    float* FO1  = alloc(1024);
    float* MH   = alloc(512);
    float* MO0  = alloc(512);
    float* MO1  = alloc(512);
    if (ws_size < off * sizeof(float)) return;   // workspace too small: bail

    // spectral buffers alias the im2col region (4 * 8.34M <= 40.2M floats)
    float* S0 = R0;
    float* S1 = R0 + SPEC_ELEMS;
    float* S2 = R0 + 2 * SPEC_ELEMS;
    float* S3 = R0 + 3 * SPEC_ELEMS;
    float* FO[2] = {FO0, FO1};
    float* MO[2] = {MO0, MO1};

    auto gemm = [&](const float* A, const float* B, float* C, const float* bias,
                    int M, int N, int K, int lda, int ldb, int ldc,
                    long long sA, long long sB, long long sC, int batch,
                    float alpha, float beta, int act) {
        dim3 g((N + 63) / 64, (M + 63) / 64, batch);
        gemm_wmma_kernel<<<g, dim3(128), 0, stream>>>(
            A, B, C, bias, M, N, K, lda, ldb, ldc, sA, sB, sC, alpha, beta, act);
    };
    auto ew = [&](long long n) { return dim3((unsigned)((n + 255) / 256)); };

    // ---- twiddles ----
    gen_wfwd_kernel<<<ew(WF * WP), 256, 0, stream>>>(TWc, TWs);
    gen_dfth_kernel<<<ew(HP * HP), 256, 0, stream>>>(Fc, Fs, -1.0f);
    gen_dfth_kernel<<<ew(HP * HP), 256, 0, stream>>>(Gc, Gs, 1.0f);
    gen_winv_kernel<<<ew(WP * WF), 256, 0, stream>>>(Ur, Ui);

    // ---- modulation embedding + per-block scale/shift MLPs ----
    mod_sincos_kernel<<<1, 32, 0, stream>>>(mod, EMB);
    matvec_kernel<<<1, 64, 0, stream>>>(emb_w, emb_b, EMB, ME, 64, 64, 1);
    matvec_kernel<<<1, 256, 0, stream>>>(modproj_w, modproj_b, ME, MV, 64, EMBED, 0);
    for (int b = 0; b < 2; ++b) {
        matvec_kernel<<<4, 256, 0, stream>>>(blk[b].fw1, blk[b].fb1, ME, FH, 64, 1024, 1);
        matvec_kernel<<<4, 256, 0, stream>>>(blk[b].fw2, blk[b].fb2, FH, FO[b], 1024, 1024, 0);
        matvec_kernel<<<2, 256, 0, stream>>>(blk[b].sw1, blk[b].sb1, ME, MH, 64, 512, 1);
        matvec_kernel<<<2, 256, 0, stream>>>(blk[b].sw2, blk[b].sb2, MH, MO[b], 512, 512, 0);
    }

    // ---- patch embed: im2col GEMM + bias + pos + modproj ----
    im2col_kernel<<<ew((long long)NTOK * KPATCH), 256, 0, stream>>>(x, R0);
    transpose_w_kernel<<<ew(KPATCH * EMBED), 256, 0, stream>>>(patch_w, BW);
    gemm(R0, BW, XT, patch_b, NTOK, EMBED, KPATCH, KPATCH, EMBED, EMBED,
         0, 0, 0, 1, 1.0f, 0.0f, 0);
    add_pos_mod_kernel<<<ew((long long)NTOK * EMBED), 256, 0, stream>>>(XT, pos, MV);

    const long long WROW = (long long)WP * EMBED;     // 360*256
    const long long FROW = (long long)WF * EMBED;     // 181*256

    for (int b = 0; b < 2; ++b) {
        const Blk& p = blk[b];
        const float* w1r = p.w1;            const float* w1i = p.w1 + 8 * 32 * 32;
        const float* w2r = p.w2;            const float* w2i = p.w2 + 8 * 32 * 32;

        // norm1
        layernorm_kernel<<<(NTOK + 7) / 8, 256, 0, stream>>>(XT, p.n1g, p.n1b, nullptr, LN, NTOK);

        // rfft along W (batched over h rows): Y = T @ X_h   [181 x 256]
        gemm(TWc, LN, S0, nullptr, WF, EMBED, WP, WP, EMBED, EMBED, 0, WROW, FROW, HP, 1.0f, 0.0f, 0);
        gemm(TWs, LN, S1, nullptr, WF, EMBED, WP, WP, EMBED, EMBED, 0, WROW, FROW, HP, 1.0f, 0.0f, 0);
        // fft along H (complex, batched over wf columns): Z = (Fc + iFs) @ Y
        gemm(Fc, S0, S2, nullptr, HP, EMBED, HP, HP, (int)FROW, (int)FROW, 0, EMBED, EMBED, WF, 1.0f, 0.0f, 0);
        gemm(Fs, S1, S2, nullptr, HP, EMBED, HP, HP, (int)FROW, (int)FROW, 0, EMBED, EMBED, WF, -1.0f, 1.0f, 0);
        gemm(Fs, S0, S3, nullptr, HP, EMBED, HP, HP, (int)FROW, (int)FROW, 0, EMBED, EMBED, WF, 1.0f, 0.0f, 0);
        gemm(Fc, S1, S3, nullptr, HP, EMBED, HP, HP, (int)FROW, (int)FROW, 0, EMBED, EMBED, WF, 1.0f, 1.0f, 0);

        // spectral layer 1: block-diagonal complex 32x32, batched over 8 blocks
        gemm(S2, w1r, S0, nullptr, SPEC_P, 32, 32, EMBED, 32, EMBED, 32, 1024, 32, 8, 1.0f, 0.0f, 0);
        gemm(S3, w1i, S0, nullptr, SPEC_P, 32, 32, EMBED, 32, EMBED, 32, 1024, 32, 8, -1.0f, 1.0f, 0);
        gemm(S3, w1r, S1, nullptr, SPEC_P, 32, 32, EMBED, 32, EMBED, 32, 1024, 32, 8, 1.0f, 0.0f, 0);
        gemm(S2, w1i, S1, nullptr, SPEC_P, 32, 32, EMBED, 32, EMBED, 32, 1024, 32, 8, 1.0f, 1.0f, 0);
        spectral_mod_kernel<<<ew((long long)SPEC_ELEMS), 256, 0, stream>>>(S0, S1, p.b1, FO[b]);

        // spectral layer 2 + softshrink
        gemm(S0, w2r, S2, nullptr, SPEC_P, 32, 32, EMBED, 32, EMBED, 32, 1024, 32, 8, 1.0f, 0.0f, 0);
        gemm(S1, w2i, S2, nullptr, SPEC_P, 32, 32, EMBED, 32, EMBED, 32, 1024, 32, 8, -1.0f, 1.0f, 0);
        gemm(S1, w2r, S3, nullptr, SPEC_P, 32, 32, EMBED, 32, EMBED, 32, 1024, 32, 8, 1.0f, 0.0f, 0);
        gemm(S0, w2i, S3, nullptr, SPEC_P, 32, 32, EMBED, 32, EMBED, 32, 1024, 32, 8, 1.0f, 1.0f, 0);
        softshrink_kernel<<<ew((long long)SPEC_ELEMS), 256, 0, stream>>>(S2, S3, p.b2);

        // inverse fft along H: Y' = (Gc + iGs) @ Z
        gemm(Gc, S2, S0, nullptr, HP, EMBED, HP, HP, (int)FROW, (int)FROW, 0, EMBED, EMBED, WF, 1.0f, 0.0f, 0);
        gemm(Gs, S3, S0, nullptr, HP, EMBED, HP, HP, (int)FROW, (int)FROW, 0, EMBED, EMBED, WF, -1.0f, 1.0f, 0);
        gemm(Gs, S2, S1, nullptr, HP, EMBED, HP, HP, (int)FROW, (int)FROW, 0, EMBED, EMBED, WF, 1.0f, 0.0f, 0);
        gemm(Gc, S3, S1, nullptr, HP, EMBED, HP, HP, (int)FROW, (int)FROW, 0, EMBED, EMBED, WF, 1.0f, 1.0f, 0);
        // inverse real DFT along W (batched over h): x = Ur@Yre + Ui@Yim -> HID
        gemm(Ur, S0, HID, nullptr, WP, EMBED, WF, WF, EMBED, EMBED, 0, FROW, WROW, HP, 1.0f, 0.0f, 0);
        gemm(Ui, S1, HID, nullptr, WP, EMBED, WF, WF, EMBED, EMBED, 0, FROW, WROW, HP, 1.0f, 1.0f, 0);

        // residuals: xt = xt + irfft + ln_input  (filter bias + double skip)
        add3_kernel<<<ew((long long)NTOK * EMBED), 256, 0, stream>>>(XT, HID, LN);

        // norm2 with mss scale/shift, then MLP with fused GELU + residual
        layernorm_kernel<<<(NTOK + 7) / 8, 256, 0, stream>>>(XT, p.n2g, p.n2b, MO[b], LN, NTOK);
        gemm(LN, p.mw1, HID, p.mb1, NTOK, LATENT, EMBED, EMBED, LATENT, LATENT, 0, 0, 0, 1, 1.0f, 0.0f, 1);
        gemm(HID, p.mw2, XT, p.mb2, NTOK, EMBED, LATENT, LATENT, EMBED, EMBED, 0, 0, 0, 1, 1.0f, 1.0f, 0);
    }

    // head + unpatchify
    gemm(XT, head_w, HEAD, nullptr, NTOK, 4 * C_OUT, EMBED, EMBED, 4 * C_OUT, 4 * C_OUT,
         0, 0, 0, 1, 1.0f, 0.0f, 0);
    unpatch_kernel<<<ew((long long)C_OUT * H_IMG * W_IMG), 256, 0, stream>>>(HEAD, (float*)d_out);
}
